// ContinuousTimeRNN_42245298324111
// MI455X (gfx1250) — matrix operand order repaired and checked by reading, not compiled, not run on hardware
//
#include <hip/hip_runtime.h>

// CDNA5 / gfx1250: wave32, WMMA 16x16x32 bf16, f32 accumulate.
typedef __attribute__((ext_vector_type(16))) __bf16 v16bf;
typedef __attribute__((ext_vector_type(8)))  __bf16 v8bf;
typedef __attribute__((ext_vector_type(8)))  float  v8f;
typedef int v4i_gcc __attribute__((vector_size(16)));   // matches async-LDS builtin

#define HDIM   512
#define NSAMP  512
#define TSTEPS 1000
#define NROWS  16      // samples (rows of h) per workgroup
#define ALPHA  0.1f

// Dynamic LDS layout (bytes). Total 321664 <= 320KB WGP limit.
#define OFF_H      0        // 16x512 f32 state            (32 KB)
#define OFF_A      32768    // 16x512 bf16 activations     (16 KB)
#define OFF_WIN0   49152    // W_in row 0                  (2 KB)
#define OFF_WIN1   51200    // W_in row 1                  (2 KB)
#define OFF_BIAS   53248    // bias                        (2 KB)
#define OFF_WOUT0  55296    // W_out col 0                 (2 KB)
#define OFF_WOUT1  57344    // W_out col 1                 (2 KB)
#define OFF_VEL    59392    // per-step velocities (16x2)  (128 B)
#define OFF_BCACHE 59520    // W_rec^T rows [0,256) bf16   (256 KB), persistent
#define SMEM_BYTES (OFF_BCACHE + 256 * HDIM * 2)

__device__ __forceinline__ unsigned short f2bf(float f) {
  unsigned int u = __builtin_bit_cast(unsigned int, f);
  u += 0x7FFFu + ((u >> 16) & 1u);
  return (unsigned short)(u >> 16);
}

// One-time: W_rec (HxH f32, row-major [k][n]) -> W_rec^T bf16 [n][k] so that
// WMMA B-fragments are contiguous 32B loads per lane.
__global__ void wrecT_bf16_kernel(const float* __restrict__ W_rec,
                                  unsigned short* __restrict__ WT) {
  int k = blockIdx.x;
  for (int n = threadIdx.x; n < HDIM; n += blockDim.x)
    WT[(size_t)n * HDIM + k] = f2bf(W_rec[(size_t)k * HDIM + n]);
}

__global__ __launch_bounds__(512)
void rnn_wmma_kernel(const float* __restrict__ initdir,   // (N,2)
                     const float* __restrict__ vel,       // (T,N,2)
                     const float* __restrict__ fc_w,      // (H,2)
                     const float* __restrict__ fc_b,      // (H)
                     const float* __restrict__ W_in,      // (2,H)
                     const float* __restrict__ W_out,     // (H,2)
                     const float* __restrict__ bias,      // (H)
                     const unsigned short* __restrict__ WTu, // (H,H) bf16, [n][k]
                     float* __restrict__ out)             // (T,N,2)
{
  extern __shared__ char smem[];
  float*          h_lds   = (float*)(smem + OFF_H);
  unsigned short* a_lds   = (unsigned short*)(smem + OFF_A);
  float*          win0    = (float*)(smem + OFF_WIN0);
  float*          win1    = (float*)(smem + OFF_WIN1);
  float*          bias_l  = (float*)(smem + OFF_BIAS);
  float*          wout0   = (float*)(smem + OFF_WOUT0);
  float*          wout1   = (float*)(smem + OFF_WOUT1);
  float*          velstep = (float*)(smem + OFF_VEL);
  unsigned short* bcache  = (unsigned short*)(smem + OFF_BCACHE);

  const int tid  = threadIdx.x;
  const int wave = tid >> 5;     // 16 waves
  const int lane = tid & 31;
  const int ln   = lane & 15;
  const int hiH  = lane >> 4;    // half-wave select (ISA fragment layouts)
  const int row0 = blockIdx.x * NROWS;

  // ---- persistent B-cache fill: W_rec^T rows [0,256) -> LDS (256 KB) ----
  // CDNA5 async global->LDS copy path (ASYNCcnt) when available.
#if defined(__HIP_DEVICE_COMPILE__) && \
    __has_builtin(__builtin_amdgcn_global_load_async_to_lds_b128)
  {
    const char* gsrc = (const char*)WTu;         // rows [0,256) are the prefix
    char*       ldst = (char*)bcache;
    for (int off = tid * 16; off < 256 * HDIM * 2; off += 512 * 16) {
      __builtin_amdgcn_global_load_async_to_lds_b128(
          (__attribute__((address_space(1))) v4i_gcc*)(gsrc + off),
          (__attribute__((address_space(3))) v4i_gcc*)(ldst + off),
          0, 0);
    }
#if __has_builtin(__builtin_amdgcn_s_wait_asynccnt)
    __builtin_amdgcn_s_wait_asynccnt(0);
#else
    asm volatile("s_wait_asynccnt 0" ::: "memory");
#endif
  }
#else
  {
    uint4*       d = (uint4*)bcache;
    const uint4* s = (const uint4*)WTu;
    for (int j = tid; j < 256 * HDIM * 2 / 16; j += 512) d[j] = s[j];
  }
#endif

  // Stage constants in LDS (reused 1000x).
  for (int i = tid; i < HDIM; i += 512) {
    win0[i]   = W_in[i];
    win1[i]   = W_in[HDIM + i];
    bias_l[i] = bias[i];
    wout0[i]  = W_out[2 * i];
    wout1[i]  = W_out[2 * i + 1];
  }
  // h0 = initdir @ fc_w^T + fc_b  (tiny: Din=2)
  for (int i = tid; i < NROWS * HDIM; i += 512) {
    int r = i >> 9, n = i & (HDIM - 1);
    int row = row0 + r;
    h_lds[i] = initdir[2 * row] * fc_w[2 * n] +
               initdir[2 * row + 1] * fc_w[2 * n + 1] + fc_b[n];
  }
  __syncthreads();

  const __bf16* WT = reinterpret_cast<const __bf16*>(WTu);
  // Each wave owns N-tiles {wave, wave+16}: columns [nA,nA+16) and [nB,nB+16).
  // Tile nA's B operand comes from the persistent LDS cache; tile nB streams L2.
  const int nA = wave << 4;
  const int nB = nA + 256;
  const __bf16* bPtrA = reinterpret_cast<const __bf16*>(bcache) +
                        (size_t)(nA + ln) * HDIM + hiH * 16;       // LDS
  const __bf16* bPtrB = WT + (size_t)(nB + ln) * HDIM + hiH * 16;  // global (L2)
  const __bf16* aPtr  = reinterpret_cast<const __bf16*>(a_lds) + ln * HDIM + hiH * 16;

  const float w0A = win0[nA + ln], w1A = win1[nA + ln], bsA = bias_l[nA + ln];
  const float w0B = win0[nB + ln], w1B = win1[nB + ln], bsB = bias_l[nB + ln];

  for (int t = 0; t < TSTEPS; ++t) {
    // ---- phase 1: a = bf16(relu(tanh(h))) on a contiguous 16-elem run ----
    // (vectorized: 4x ds_load_b128 + 2x ds_store_b128 per thread per step)
    {
      const int base = tid * 16;               // 512 threads x 16 = 8192 elems
      float hv[16];
#pragma unroll
      for (int j = 0; j < 4; ++j)
        *(float4*)&hv[4 * j] = *(const float4*)(h_lds + base + 4 * j);
      unsigned int w[8];
#pragma unroll
      for (int j = 0; j < 8; ++j) {
        float a0 = tanhf(hv[2 * j]);
        float a1 = tanhf(hv[2 * j + 1]);
        a0 = a0 > 0.f ? a0 : 0.f;
        a1 = a1 > 0.f ? a1 : 0.f;
        w[j] = (unsigned int)f2bf(a0) | ((unsigned int)f2bf(a1) << 16);
      }
      *(uint4*)(a_lds + base)     = make_uint4(w[0], w[1], w[2], w[3]);
      *(uint4*)(a_lds + base + 8) = make_uint4(w[4], w[5], w[6], w[7]);
    }
    if (tid < NROWS * 2)
      velstep[tid] = vel[((size_t)t * NSAMP + row0 + (tid >> 1)) * 2 + (tid & 1)];
    __syncthreads();

    // ---- phase 2: (16x512) @ (512x512) slice via bf16 WMMA, K-loop of 16 ----
    v8f accA = {0.f, 0.f, 0.f, 0.f, 0.f, 0.f, 0.f, 0.f};
    v8f accB = accA;
#pragma unroll 4
    for (int k0 = 0; k0 < HDIM; k0 += 32) {
      // A fragment (16x32 bf16): lane ln = row m; contiguous K=k0+hiH*16..+15
      // (same k order as the B fragments)
      v8bf alo = *reinterpret_cast<const v8bf*>(aPtr + k0);
      v8bf ahi = *reinterpret_cast<const v8bf*>(aPtr + k0 + 8);
      v16bf afrag = __builtin_shufflevector(alo, ahi,
          0,1,2,3,4,5,6,7,8,9,10,11,12,13,14,15);
      // B fragments (32x16 bf16): lane holds column n, contiguous K=k0+hiH*16..+15
      v8bf bAlo = *reinterpret_cast<const v8bf*>(bPtrA + k0);      // ds_load_b128
      v8bf bAhi = *reinterpret_cast<const v8bf*>(bPtrA + k0 + 8);
      v16bf bfA = __builtin_shufflevector(bAlo, bAhi,
          0,1,2,3,4,5,6,7,8,9,10,11,12,13,14,15);
      v8bf bBlo = *reinterpret_cast<const v8bf*>(bPtrB + k0);      // global_load_b128
      v8bf bBhi = *reinterpret_cast<const v8bf*>(bPtrB + k0 + 8);
      v16bf bfB = __builtin_shufflevector(bBlo, bBhi,
          0,1,2,3,4,5,6,7,8,9,10,11,12,13,14,15);
      // WGP-scope (default-scope) prefetch ~8 slabs ahead so demand loads hit
      // the near cache; SE/SYS scope (all __builtin_prefetch gives) only warms
      // GL2 where W_rec^T already resides.
      asm volatile("global_prefetch_b8 %0, off offset:512"
                   :: "v"((const void*)(bPtrB + k0)));
      accA = __builtin_amdgcn_wmma_f32_16x16x32_bf16(
          false, afrag, false, bfA, (short)0, accA, false, false);
      accB = __builtin_amdgcn_wmma_f32_16x16x32_bf16(
          false, afrag, false, bfB, (short)0, accB, false, false);
    }

    // ---- fused state update: h = 0.9h + 0.1(acc + x@W_in + b) ----
    // C/D layout: VGPR v holds row m = v + 8*hiH, column n = tile + ln
#pragma unroll
    for (int v = 0; v < 8; ++v) {
      int m = hiH * 8 + v;
      float x0 = velstep[2 * m], x1 = velstep[2 * m + 1];
      int ia = m * HDIM + nA + ln;
      int ib = m * HDIM + nB + ln;
      h_lds[ia] = (1.0f - ALPHA) * h_lds[ia] + ALPHA * (accA[v] + x0 * w0A + x1 * w1A + bsA);
      h_lds[ib] = (1.0f - ALPHA) * h_lds[ib] + ALPHA * (accB[v] + x0 * w0B + x1 * w1B + bsB);
    }
    __syncthreads();

    // ---- phase 3: out[t, row, :] = h_new[row, :] @ W_out  (wave per row) ----
    {
      const float* hr = h_lds + wave * HDIM;
      float s0 = 0.f, s1 = 0.f;
#pragma unroll
      for (int k = lane; k < HDIM; k += 32) {
        float hv = hr[k];
        s0 += hv * wout0[k];
        s1 += hv * wout1[k];
      }
#pragma unroll
      for (int mask = 16; mask >= 1; mask >>= 1) {
        s0 += __shfl_xor(s0, mask, 32);
        s1 += __shfl_xor(s1, mask, 32);
      }
      if (lane == 0) {
        size_t o = ((size_t)t * NSAMP + row0 + wave) * 2;
        *(float2*)(out + o) = make_float2(s0, s1);
      }
    }
  }
}

extern "C" void kernel_launch(void* const* d_in, const int* in_sizes, int n_in,
                              void* d_out, int out_size, void* d_ws, size_t ws_size,
                              hipStream_t stream) {
  const float* initdir = (const float*)d_in[0];
  const float* vel     = (const float*)d_in[1];
  const float* fc_w    = (const float*)d_in[2];
  const float* fc_b    = (const float*)d_in[3];
  const float* W_in    = (const float*)d_in[4];
  const float* W_rec   = (const float*)d_in[5];
  const float* W_out   = (const float*)d_in[6];
  const float* bias    = (const float*)d_in[7];
  unsigned short* WT = (unsigned short*)d_ws;   // 512*512*2 = 512 KB scratch
  float* out = (float*)d_out;

  // allow >64KB dynamic LDS (WGP supports 320KB per workgroup on gfx1250)
  (void)hipFuncSetAttribute((const void*)rnn_wmma_kernel,
                            hipFuncAttributeMaxDynamicSharedMemorySize,
                            SMEM_BYTES);

  hipLaunchKernelGGL(wrecT_bf16_kernel, dim3(HDIM), dim3(256), 0, stream, W_rec, WT);
  hipLaunchKernelGGL(rnn_wmma_kernel, dim3(NSAMP / NROWS), dim3(512), SMEM_BYTES, stream,
                     initdir, vel, fc_w, fc_b, W_in, W_out, bias, WT, out);
}
